// PointNetAlphaUNet_7954279432843
// MI455X (gfx1250) — compile-verified, matrix operand order checked by reading, not run on hardware
//
#include <hip/hip_runtime.h>

// ---------------------------------------------------------------------------
// Types for CDNA5 WMMA (gfx1250, wave32)
// ---------------------------------------------------------------------------
typedef __attribute__((ext_vector_type(16))) __bf16 v16bf;
typedef __attribute__((ext_vector_type(8)))  float  v8f;

union Frag16 { v16bf v; unsigned int u[8]; };

// ---------------------------------------------------------------------------
// WMMA GEMM:  out[M,N] = act( A[M,Kpad](bf16) @ W^T[N,Kpad](bf16) + bias[N] )
// Block = 128 threads (4 waves); each wave computes one 16x16 tile.
// M multiple of 16, N multiple of 64, Kpad multiple of 32. No divergence ->
// EXEC stays all-ones around v_wmma as the ISA requires.
//
// 16-bit A fragment layout (ISA 7.12.2): lane l holds row m = l&15; half =
// l>>4 selects K sub-range; pair v holds k = (v<4 ? 2v : 2v+8) + 8*half.
// B fragment mirrors with N<->M; C/D: lane owns column n=l&15, rows v+8*half.
// ---------------------------------------------------------------------------
template <typename OutT>
__global__ void gemm_wmma_kernel(const __bf16* __restrict__ A,
                                 const __bf16* __restrict__ Bt,
                                 const float*  __restrict__ bias,
                                 OutT* __restrict__ out,
                                 int M, int K, int Nn, int relu) {
  const int lane = threadIdx.x & 31;
  const int wave = threadIdx.x >> 5;
  const int tm   = blockIdx.x;
  const int tn   = blockIdx.y * 4 + wave;
  const int half = lane >> 4;
  const int mrow = tm * 16 + (lane & 15);
  const int ncol = tn * 16 + (lane & 15);

  const unsigned int* __restrict__ Arow =
      (const unsigned int*)(A  + (size_t)mrow * K);
  const unsigned int* __restrict__ Brow =
      (const unsigned int*)(Bt + (size_t)ncol * K);

  v8f acc = {};
  for (int k0 = 0; k0 < K; k0 += 32) {
    Frag16 a, b;
#pragma unroll
    for (int v = 0; v < 8; ++v) {
      const int kk = k0 + ((v < 4) ? (2 * v) : (2 * v + 8)) + 8 * half;
      a.u[v] = Arow[kk >> 1];
      b.u[v] = Brow[kk >> 1];
    }
    acc = __builtin_amdgcn_wmma_f32_16x16x32_bf16(
        /*neg_a=*/false, a.v, /*neg_b=*/false, b.v,
        /*c_mod=*/(short)0, acc, /*reuse_a=*/false, /*reuse_b=*/false);
  }

  const float bn = bias[ncol];
#pragma unroll
  for (int v = 0; v < 8; ++v) {
    const int r = tm * 16 + v + 8 * half;
    float val = acc[v] + bn;
    if (relu) val = fmaxf(val, 0.0f);
    out[(size_t)r * Nn + ncol] = (OutT)val;
  }
}

// ---------------------------------------------------------------------------
// Weight convert+transpose: W[K,N] f32  ->  Wt[N,Kpad] bf16 (zero padded)
// ---------------------------------------------------------------------------
__global__ void convert_wt_kernel(const float* __restrict__ W,
                                  __bf16* __restrict__ Wt,
                                  int K, int Nn, int Kpad) {
  const int gid = blockIdx.x * 256 + threadIdx.x;
  const int total = Nn * Kpad;
  if (gid >= total) return;
  const int kp = gid % Kpad;
  const int n  = gid / Kpad;
  const float v = (kp < K) ? W[(size_t)kp * Nn + n] : 0.0f;
  Wt[(size_t)n * Kpad + kp] = (__bf16)v;
}

// ---------------------------------------------------------------------------
// Pack/concat up to 3 f32 row sources into zero-padded bf16 [M,Kpad]
// ---------------------------------------------------------------------------
__global__ void pack_kernel(const float* __restrict__ s0, int w0,
                            const float* __restrict__ s1, int w1,
                            const float* __restrict__ s2, int w2,
                            __bf16* __restrict__ dst,
                            long long total, int Kpad) {
  const long long gid = (long long)blockIdx.x * 256 + threadIdx.x;
  if (gid >= total) return;
  const int c       = (int)(gid % Kpad);
  const long long r = gid / Kpad;
  float v = 0.0f;
  if (c < w0)                       v = s0[r * w0 + c];
  else if (c < w0 + w1)             v = s1[r * w1 + (c - w0)];
  else if (s2 && c < w0 + w1 + w2)  v = s2[r * w2 + (c - w0 - w1)];
  dst[gid] = (__bf16)v;
}

// ---------------------------------------------------------------------------
// Farthest point sampling, one block (256 thr) per cloud; min-d2 in registers.
// NE = N/256. Writes sampled positions directly.
// ---------------------------------------------------------------------------
template <int NE>
__global__ void fps_kernel(const float* __restrict__ pos, int N, int nS,
                           float* __restrict__ outpos) {
  const int b = blockIdx.x;
  const int t = threadIdx.x;
  const float* __restrict__ P = pos + (size_t)b * N * 3;

  __shared__ float rv[256];
  __shared__ int   ri[256];
  __shared__ float lastp[3];

  float mind[NE];
#pragma unroll
  for (int e = 0; e < NE; ++e) mind[e] = 3.4e38f;

  if (t == 0) {
    lastp[0] = P[0]; lastp[1] = P[1]; lastp[2] = P[2];
    float* op = outpos + (size_t)b * nS * 3;
    op[0] = P[0]; op[1] = P[1]; op[2] = P[2];
  }
  __syncthreads();

  for (int s = 1; s < nS; ++s) {
    const float lx = lastp[0], ly = lastp[1], lz = lastp[2];
    float bestd = -1.0f; int besti = 0;
#pragma unroll
    for (int e = 0; e < NE; ++e) {
      const int i = t + e * 256;
      const float dx = P[i * 3 + 0] - lx;
      const float dy = P[i * 3 + 1] - ly;
      const float dz = P[i * 3 + 2] - lz;
      const float d2 = dx * dx + dy * dy + dz * dz;
      mind[e] = fminf(mind[e], d2);
      if (mind[e] > bestd) { bestd = mind[e]; besti = i; }
    }
    rv[t] = bestd; ri[t] = besti;
    __syncthreads();
    for (int off = 128; off > 0; off >>= 1) {
      if (t < off && rv[t + off] > rv[t]) { rv[t] = rv[t + off]; ri[t] = ri[t + off]; }
      __syncthreads();
    }
    if (t == 0) {
      const int bi = ri[0];
      const float px = P[bi * 3 + 0], py = P[bi * 3 + 1], pz = P[bi * 3 + 2];
      lastp[0] = px; lastp[1] = py; lastp[2] = pz;
      float* op = outpos + ((size_t)b * nS + s) * 3;
      op[0] = px; op[1] = py; op[2] = pz;
    }
    __syncthreads();
  }
}

// ---------------------------------------------------------------------------
// kNN search: one thread per target; unrolled sorted top-K insertion.
// Emits source indices, weights 1/max(d2,1e-16) and weight sum (PyG semantics).
// ---------------------------------------------------------------------------
template <int K>
__global__ void knn_search_kernel(const float* __restrict__ tpos,
                                  const float* __restrict__ spos,
                                  int Nt, int Ns,
                                  int*   __restrict__ oidx,
                                  float* __restrict__ ow,
                                  float* __restrict__ owsum,
                                  int total) {
  const int gid = blockIdx.x * blockDim.x + threadIdx.x;
  if (gid >= total) return;
  const int b = gid / Nt;
  const float tx = tpos[(size_t)gid * 3 + 0];
  const float ty = tpos[(size_t)gid * 3 + 1];
  const float tz = tpos[(size_t)gid * 3 + 2];
  const float* __restrict__ S = spos + (size_t)b * Ns * 3;

  float bd[K]; int bi[K];
#pragma unroll
  for (int q = 0; q < K; ++q) { bd[q] = 3.4e38f; bi[q] = 0; }

  for (int j = 0; j < Ns; ++j) {
    __builtin_prefetch(&S[(j + 64) * 3], 0, 1);   // global_prefetch_b8
    const float dx = S[j * 3 + 0] - tx;
    const float dy = S[j * 3 + 1] - ty;
    const float dz = S[j * 3 + 2] - tz;
    const float d2 = dx * dx + dy * dy + dz * dz;
    if (d2 < bd[K - 1]) {
      bd[K - 1] = d2; bi[K - 1] = j;
#pragma unroll
      for (int q = K - 1; q > 0; --q) {
        if (bd[q] < bd[q - 1]) {
          const float td = bd[q]; bd[q] = bd[q - 1]; bd[q - 1] = td;
          const int   ti = bi[q]; bi[q] = bi[q - 1]; bi[q - 1] = ti;
        }
      }
    }
  }
  float wsum = 0.0f;
#pragma unroll
  for (int q = 0; q < K; ++q) {
    const float w = 1.0f / fmaxf(bd[q], 1e-16f);
    ow[(size_t)gid * K + q]   = w;
    oidx[(size_t)gid * K + q] = bi[q];
    wsum += w;
  }
  owsum[gid] = wsum;
}

// ---------------------------------------------------------------------------
// Weighted gather: out[t,c] = sum_k w[t,k] * feat[b, idx[t,k], c] / wsum[t]
// One thread per (target, channel) -> coalesced on c.
// ---------------------------------------------------------------------------
template <int K>
__global__ void knn_gather_kernel(const float* __restrict__ feat,
                                  const int*   __restrict__ idx,
                                  const float* __restrict__ w,
                                  const float* __restrict__ wsum,
                                  float* __restrict__ out,
                                  long long total, int Nt, int Ns, int C) {
  const long long gid = (long long)blockIdx.x * 256 + threadIdx.x;
  if (gid >= total) return;
  const int c       = (int)(gid % C);
  const long long t = gid / C;
  const int b       = (int)(t / Nt);
  float acc = 0.0f;
#pragma unroll
  for (int q = 0; q < K; ++q) {
    const int j = idx[t * K + q];
    acc += w[t * K + q] * feat[((size_t)b * Ns + j) * C + c];
  }
  out[gid] = acc / wsum[t];
}

// ---------------------------------------------------------------------------
// Head: mean = H[M,64] @ w2[64] + b2; out = softplus(mean) + 0.01
// ---------------------------------------------------------------------------
__global__ void head_out_kernel(const float* __restrict__ H,
                                const float* __restrict__ w2,
                                const float* __restrict__ b2,
                                float* __restrict__ out, int M) {
  const int gid = blockIdx.x * 256 + threadIdx.x;
  if (gid >= M) return;
  float v = b2[0];
  const float* __restrict__ h = H + (size_t)gid * 64;
#pragma unroll
  for (int k = 0; k < 64; ++k) v += h[k] * w2[k];
  const float sp = (v > 20.0f) ? v : log1pf(expf(v));
  out[gid] = sp + 0.01f;
}

// ---------------------------------------------------------------------------
// Host orchestration
// ---------------------------------------------------------------------------
extern "C" void kernel_launch(void* const* d_in, const int* in_sizes, int n_in,
                              void* d_out, int out_size, void* d_ws, size_t ws_size,
                              hipStream_t stream) {
  (void)in_sizes; (void)n_in; (void)out_size; (void)ws_size;
  const int B = 8, N = 8192, N1 = 2048, N2 = 512;
  const int M0 = B * N, M1 = B * N1, M2 = B * N2;

  const float* pos = (const float*)d_in[0];
  const float* x   = (const float*)d_in[1];
  auto P = [&](int i) { return (const float*)d_in[i]; };
  // param order: sa1(w1,b1,w2,b2) sa2 sa3 fp2 fp1 fp0 head(w1,b1,w2,b2)
  enum { SA1_W1 = 2, SA1_B1, SA1_W2, SA1_B2,
         SA2_W1,     SA2_B1, SA2_W2, SA2_B2,
         SA3_W1,     SA3_B1, SA3_W2, SA3_B2,
         FP2_W1,     FP2_B1, FP2_W2, FP2_B2,
         FP1_W1,     FP1_B1, FP1_W2, FP1_B2,
         FP0_W1,     FP0_B1, FP0_W2, FP0_B2,
         HD_W1,      HD_B1,  HD_W2,  HD_B2 };

  char* ws = (char*)d_ws;
  size_t off = 0;
  auto alloc = [&](size_t bytes) -> char* {
    char* p = ws + off; off += (bytes + 255) & ~(size_t)255; return p;
  };

  __bf16* A_arena = (__bf16*)alloc((size_t)M0 * 384 * 2); // packed activations
  __bf16* H_arena = (__bf16*)alloc((size_t)M0 * 256 * 2); // MLP hidden
  float*  G_arena = (float*) alloc((size_t)M0 * 256 * 4); // gather outputs / reuse
  float*  F0      = (float*) alloc((size_t)M0 * 128 * 4); // l0_feat, later "final"
  float*  l1_feat = (float*) alloc((size_t)M1 * 256 * 4);
  float*  l2_feat = (float*) alloc((size_t)M2 * 512 * 4);
  float*  l1_fp   = (float*) alloc((size_t)M1 * 256 * 4);
  float*  l1_pos  = (float*) alloc((size_t)M1 * 3 * 4);
  float*  l2_pos  = (float*) alloc((size_t)M2 * 3 * 4);
  int*    kidx    = (int*)   alloc((size_t)M1 * 16 * 4);  // >= M0*3
  float*  kw      = (float*) alloc((size_t)M1 * 16 * 4);
  float*  kws     = (float*) alloc((size_t)M0 * 4);

  __bf16* wt_sa1_1 = (__bf16*)alloc((size_t)64  * 32  * 2);
  __bf16* wt_sa1_2 = (__bf16*)alloc((size_t)128 * 64  * 2);
  __bf16* wt_sa2_1 = (__bf16*)alloc((size_t)128 * 160 * 2);
  __bf16* wt_sa2_2 = (__bf16*)alloc((size_t)256 * 128 * 2);
  __bf16* wt_sa3_1 = (__bf16*)alloc((size_t)256 * 288 * 2);
  __bf16* wt_sa3_2 = (__bf16*)alloc((size_t)512 * 256 * 2);
  __bf16* wt_fp2_1 = (__bf16*)alloc((size_t)256 * 768 * 2);
  __bf16* wt_fp2_2 = (__bf16*)alloc((size_t)256 * 256 * 2);
  __bf16* wt_fp1_1 = (__bf16*)alloc((size_t)256 * 384 * 2);
  __bf16* wt_fp1_2 = (__bf16*)alloc((size_t)128 * 256 * 2);
  __bf16* wt_fp0_1 = (__bf16*)alloc((size_t)128 * 160 * 2);
  __bf16* wt_fp0_2 = (__bf16*)alloc((size_t)128 * 128 * 2);
  __bf16* wt_head1 = (__bf16*)alloc((size_t)64  * 128 * 2);

  auto cvt = [&](const float* src, __bf16* dst, int K, int Nn, int Kpad) {
    const int total = Nn * Kpad;
    convert_wt_kernel<<<dim3((total + 255) / 256), dim3(256), 0, stream>>>(
        src, dst, K, Nn, Kpad);
  };
  auto pack = [&](const float* s0, int w0, const float* s1, int w1,
                  const float* s2, int w2, __bf16* dst, int M, int Kpad) {
    const long long total = (long long)M * Kpad;
    pack_kernel<<<dim3((unsigned)((total + 255) / 256)), dim3(256), 0, stream>>>(
        s0, w0, s1, w1, s2, w2, dst, total, Kpad);
  };
  auto gemm_bf = [&](const __bf16* A, const __bf16* Bt, const float* bias,
                     __bf16* out, int M, int K, int Nn) {
    gemm_wmma_kernel<__bf16><<<dim3(M / 16, Nn / 64), dim3(128), 0, stream>>>(
        A, Bt, bias, out, M, K, Nn, 1);
  };
  auto gemm_f = [&](const __bf16* A, const __bf16* Bt, const float* bias,
                    float* out, int M, int K, int Nn) {
    gemm_wmma_kernel<float><<<dim3(M / 16, Nn / 64), dim3(128), 0, stream>>>(
        A, Bt, bias, out, M, K, Nn, 1);
  };

  // --- weights -> bf16 [N, Kpad] ---
  cvt(P(SA1_W1), wt_sa1_1, 6,   64,  32);
  cvt(P(SA1_W2), wt_sa1_2, 64,  128, 64);
  cvt(P(SA2_W1), wt_sa2_1, 131, 128, 160);
  cvt(P(SA2_W2), wt_sa2_2, 128, 256, 128);
  cvt(P(SA3_W1), wt_sa3_1, 259, 256, 288);
  cvt(P(SA3_W2), wt_sa3_2, 256, 512, 256);
  cvt(P(FP2_W1), wt_fp2_1, 768, 256, 768);
  cvt(P(FP2_W2), wt_fp2_2, 256, 256, 256);
  cvt(P(FP1_W1), wt_fp1_1, 384, 256, 384);
  cvt(P(FP1_W2), wt_fp1_2, 256, 128, 256);
  cvt(P(FP0_W1), wt_fp0_1, 134, 128, 160);
  cvt(P(FP0_W2), wt_fp0_2, 128, 128, 128);
  cvt(P(HD_W1),  wt_head1, 128, 64,  128);

  // --- sa1: l0_feat = MLP([pos,x]) ---
  pack(pos, 3, x, 3, nullptr, 0, A_arena, M0, 32);
  gemm_bf(A_arena, wt_sa1_1, P(SA1_B1), H_arena, M0, 32, 64);
  gemm_f (H_arena, wt_sa1_2, P(SA1_B2), F0,      M0, 64, 128);

  // --- FPS level 1: pos -> l1_pos ---
  fps_kernel<32><<<dim3(B), dim3(256), 0, stream>>>(pos, N, N1, l1_pos);

  // --- kNN(16) down 1: l1_agg(G) = interp(l0_feat, pos -> l1_pos) ---
  knn_search_kernel<16><<<dim3((M1 + 63) / 64), dim3(64), 0, stream>>>(
      l1_pos, pos, N1, N, kidx, kw, kws, M1);
  {
    const long long total = (long long)M1 * 128;
    knn_gather_kernel<16><<<dim3((unsigned)((total + 255) / 256)), dim3(256), 0, stream>>>(
        F0, kidx, kw, kws, G_arena, total, N1, N, 128);
  }

  // --- sa2: l1_feat = MLP([l1_agg, l1_pos]) ---
  pack(G_arena, 128, l1_pos, 3, nullptr, 0, A_arena, M1, 160);
  gemm_bf(A_arena, wt_sa2_1, P(SA2_B1), H_arena, M1, 160, 128);
  gemm_f (H_arena, wt_sa2_2, P(SA2_B2), l1_feat, M1, 128, 256);

  // --- FPS level 2: l1_pos -> l2_pos ---
  fps_kernel<8><<<dim3(B), dim3(256), 0, stream>>>(l1_pos, N1, N2, l2_pos);

  // --- kNN(16) down 2: l2_agg(G) = interp(l1_feat, l1_pos -> l2_pos) ---
  knn_search_kernel<16><<<dim3((M2 + 63) / 64), dim3(64), 0, stream>>>(
      l2_pos, l1_pos, N2, N1, kidx, kw, kws, M2);
  {
    const long long total = (long long)M2 * 256;
    knn_gather_kernel<16><<<dim3((unsigned)((total + 255) / 256)), dim3(256), 0, stream>>>(
        l1_feat, kidx, kw, kws, G_arena, total, N2, N1, 256);
  }

  // --- sa3: l2_feat = MLP([l2_agg, l2_pos]) ---
  pack(G_arena, 256, l2_pos, 3, nullptr, 0, A_arena, M2, 288);
  gemm_bf(A_arena, wt_sa3_1, P(SA3_B1), H_arena, M2, 288, 256);
  gemm_f (H_arena, wt_sa3_2, P(SA3_B2), l2_feat, M2, 256, 512);

  // --- up 1: l1_up(G) = interp3(l2_feat, l2_pos -> l1_pos); fp2 MLP ---
  knn_search_kernel<3><<<dim3((M1 + 63) / 64), dim3(64), 0, stream>>>(
      l1_pos, l2_pos, N1, N2, kidx, kw, kws, M1);
  {
    const long long total = (long long)M1 * 512;
    knn_gather_kernel<3><<<dim3((unsigned)((total + 255) / 256)), dim3(256), 0, stream>>>(
        l2_feat, kidx, kw, kws, G_arena, total, N1, N2, 512);
  }
  pack(G_arena, 512, l1_feat, 256, nullptr, 0, A_arena, M1, 768);
  gemm_bf(A_arena, wt_fp2_1, P(FP2_B1), H_arena, M1, 768, 256);
  gemm_f (H_arena, wt_fp2_2, P(FP2_B2), l1_fp,   M1, 256, 256);

  // --- up 0: l0_up(G) = interp3(l1_fp, l1_pos -> pos); fp1 MLP ---
  knn_search_kernel<3><<<dim3((M0 + 63) / 64), dim3(64), 0, stream>>>(
      pos, l1_pos, N, N1, kidx, kw, kws, M0);
  {
    const long long total = (long long)M0 * 256;
    knn_gather_kernel<3><<<dim3((unsigned)((total + 255) / 256)), dim3(256), 0, stream>>>(
        l1_fp, kidx, kw, kws, G_arena, total, N, N1, 256);
  }
  pack(G_arena, 256, F0, 128, nullptr, 0, A_arena, M0, 384); // [l0_up, l0_feat]
  gemm_bf(A_arena, wt_fp1_1, P(FP1_B1), H_arena, M0, 384, 256);
  float* l0_fp = G_arena;                 // l0_up dead after pack -> reuse
  gemm_f (H_arena, wt_fp1_2, P(FP1_B2), l0_fp,   M0, 256, 128);

  // --- fp0: final(F0) = MLP([l0_fp, pos, x]) ---
  pack(l0_fp, 128, pos, 3, x, 3, A_arena, M0, 160);
  gemm_bf(A_arena, wt_fp0_1, P(FP0_B1), H_arena, M0, 160, 128);
  float* finalb = F0;                     // l0_feat dead -> reuse
  gemm_f (H_arena, wt_fp0_2, P(FP0_B2), finalb,  M0, 128, 128);

  // --- head: relu(final @ w1 + b1) via WMMA, then scalar 64->1 + softplus ---
  pack(finalb, 128, nullptr, 0, nullptr, 0, A_arena, M0, 128);
  float* Hhead = G_arena;                 // l0_fp dead -> reuse
  gemm_f(A_arena, wt_head1, P(HD_B1), Hhead, M0, 128, 64);
  head_out_kernel<<<dim3((M0 + 255) / 256), dim3(256), 0, stream>>>(
      Hhead, P(HD_W2), P(HD_B2), (float*)d_out, M0);
}
